// EdgeGATLayer_3719441679128
// MI455X (gfx1250) — matrix operand order repaired and checked by reading, not compiled, not run on hardware
//
#include <hip/hip_runtime.h>

// ---------------------------------------------------------------------------
// EdgeGAT layer for MI455X (gfx1250): all GEMMs via v_wmma_f32_16x16x32_bf16,
// attention K/V staging via the Tensor Data Mover (tensor_load_to_lds).
// D=256 tokens-per-axis, 65536 edge tokens of dim 128, H=4 heads of HD=32.
// ---------------------------------------------------------------------------

typedef __attribute__((ext_vector_type(16))) __bf16       v16bf;
typedef __attribute__((ext_vector_type(8)))  float        v8f;
typedef __attribute__((ext_vector_type(4)))  unsigned int u32x4;
typedef __attribute__((ext_vector_type(8)))  unsigned int u32x8;

#define DD   256
#define CIN  128
#define COUT 128
#define NH   4
#define HD   32
#define NTOK (DD * DD)

union FragU {
    uint4  u4[2];
    __bf16 h[16];
    v16bf  v;
};

union V32U {
    uint4  u4[4];
    __bf16 h[32];
};

// Raw LDS byte offset of a __shared__ object (TDM D# wants LDS offsets).
__device__ __forceinline__ unsigned lds_offset(void* p) {
    return (unsigned)(unsigned long long)(__attribute__((address_space(3))) char*)p;
}

// ---------------------------------------------------------------------------
// TDM: 2D tile load Global -> LDS. `rows` lines of `row_elems` bf16 elements,
// consecutive lines `row_stride_elems` apart; packed contiguously in LDS.
// D# layout per CDNA5 ISA 8.3/8.4 (2D tensor: groups 2/3 unused).
// Issue from ONE wave only; completion via s_wait_tensorcnt.
// ---------------------------------------------------------------------------
__device__ __forceinline__ void tdm_load_2d(unsigned lds_off, const void* gptr,
                                            unsigned rows, unsigned row_elems,
                                            unsigned row_stride_elems) {
    unsigned long long ga = (unsigned long long)gptr;
    u32x4 g0;
    g0[0] = 1u;                                   // count=1 (valid user D#)
    g0[1] = lds_off;                              // lds_addr (bytes)
    g0[2] = (unsigned)(ga & 0xffffffffu);         // global_addr[31:0]
    g0[3] = (unsigned)((ga >> 32) & 0x01ffffffu)  // global_addr[56:32]
          | (2u << 30);                           // type = 2 ("image")
    u32x8 g1;
    g1[0] = 1u << 16;                 // wg_mask=0 | data_size=1 (2 bytes)
    g1[1] = row_elems << 16;          // barrier_addr=0 | tensor_dim0[15:0]
    g1[2] = rows << 16;               // tensor_dim0[31:16]=0 | tensor_dim1[15:0]
    g1[3] = row_elems << 16;          // tensor_dim1[31:16]=0 | tile_dim0
    g1[4] = rows;                     // tile_dim1 | tile_dim2=0
    g1[5] = row_stride_elems;         // tensor_dim0_stride[31:0]
    g1[6] = 0u;                       // stride[47:32]=0 | dim1_stride lo (unused 2D)
    g1[7] = 0u;
    asm volatile("tensor_load_to_lds %0, %1" :: "s"(g0), "s"(g1) : "memory");
}

// A-fragment (16x32 bf16, M x K). src row-major [m][k], row stride ld (elems).
// lane m = lane&15; kb = (lane<16?0:8); element i -> k = kb + i + (i&8).
__device__ __forceinline__ v16bf load_frag_a(const __bf16* src, int ld, int lane) {
    int m  = lane & 15;
    int kb = (lane < 16) ? 0 : 8;
    FragU f;
    f.u4[0] = *(const uint4*)(src + (size_t)m * ld + kb);       // i = 0..7
    f.u4[1] = *(const uint4*)(src + (size_t)m * ld + kb + 16);  // i = 8..15
    return f.v;
}

// B-fragment (32x16 bf16, K x N) from B^T row-major: bT[n][k], stride ld.
// lane n = lane&15; kb = (lane<16?0:8); element i -> k = kb + i + (i&8).
__device__ __forceinline__ v16bf load_frag_bT(const __bf16* bT, int ld, int lane) {
    int n  = lane & 15;
    int kb = (lane < 16) ? 0 : 8;
    FragU f;
    f.u4[0] = *(const uint4*)(bT + (size_t)n * ld + kb);
    f.u4[1] = *(const uint4*)(bT + (size_t)n * ld + kb + 16);
    return f.v;
}

__device__ __forceinline__ v8f wmma_bf16(v16bf a, v16bf b, v8f c) {
    return __builtin_amdgcn_wmma_f32_16x16x32_bf16(
        /*neg_a=*/false, a, /*neg_b=*/false, b,
        /*c_mod=*/(short)0, c, /*reuse_a=*/false, /*reuse_b=*/false);
}

// ---------------------------------------------------------------------------
// Kernel 1: Q/K/V projection. grid = (NTOK/128, 3), block = 256 (8 waves).
// C[tok,128] = bf16( fp32(edge_feat[tok,:]) @ W ), W selected by blockIdx.y.
// (fp32 -> bf16 conversion during staging, so plain loads rather than TDM.)
// ---------------------------------------------------------------------------
__global__ __launch_bounds__(256)
void qkv_proj_kernel(const float* __restrict__ ef,
                     const float* __restrict__ Wq,
                     const float* __restrict__ Wk,
                     const float* __restrict__ Wv,
                     __bf16* __restrict__ Qb,
                     __bf16* __restrict__ Kb,
                     __bf16* __restrict__ Vb) {
    __shared__ __bf16 sWt[COUT][CIN];   // W^T : sWt[n][k]  (32KB)
    __shared__ __bf16 sA [128][CIN];    // A tile row-major (32KB)

    const int which = blockIdx.y;
    const float* W   = (which == 0) ? Wq : (which == 1) ? Wk : Wv;
    __bf16*      Out = (which == 0) ? Qb : (which == 1) ? Kb : Vb;

    const int    t    = threadIdx.x;
    const size_t tok0 = (size_t)blockIdx.x * 128;

    {
        const int r  = t >> 1;
        const int c0 = (t & 1) * 64;
        const float* wsrc = W + r * COUT + c0;
        #pragma unroll
        for (int j = 0; j < 64; j += 4) {
            float4 f = *(const float4*)(wsrc + j);
            sWt[c0 + j + 0][r] = (__bf16)f.x;
            sWt[c0 + j + 1][r] = (__bf16)f.y;
            sWt[c0 + j + 2][r] = (__bf16)f.z;
            sWt[c0 + j + 3][r] = (__bf16)f.w;
        }
        const float* asrc = ef + (tok0 + r) * CIN + c0;
        #pragma unroll
        for (int j = 0; j < 64; j += 4) {
            float4 f = *(const float4*)(asrc + j);
            sA[r][c0 + j + 0] = (__bf16)f.x;
            sA[r][c0 + j + 1] = (__bf16)f.y;
            sA[r][c0 + j + 2] = (__bf16)f.z;
            sA[r][c0 + j + 3] = (__bf16)f.w;
        }
    }
    __syncthreads();

    const int lane = t & 31;
    const int wave = t >> 5;
    const int half = (lane < 16) ? 0 : 8;

    #pragma unroll
    for (int nt = 0; nt < 8; ++nt) {
        v8f acc = {};
        #pragma unroll
        for (int kt = 0; kt < 4; ++kt) {
            v16bf a = load_frag_a (&sA [wave * 16][kt * 32], CIN, lane);
            v16bf b = load_frag_bT(&sWt[nt * 16 ][kt * 32], CIN, lane);
            acc = wmma_bf16(a, b, acc);
        }
        const int n = nt * 16 + (lane & 15);
        #pragma unroll
        for (int g = 0; g < 8; ++g) {
            size_t row = tok0 + wave * 16 + g + half;
            Out[row * COUT + n] = (__bf16)acc[g];
        }
    }
}

// ---------------------------------------------------------------------------
// Kernel 2: axial attention for one (index c, head h, q-half).
// mode 0: column attention (tokens q*256+c), stores 0.5*out.
// mode 1: row attention    (tokens c*256+q), accumulates += 0.5*out.
// grid = (256, 4, 2), block = 256 (8 waves; each wave owns 16 q-rows).
// K/V tiles fetched by the Tensor Data Mover (one strided 2D DMA each).
// ---------------------------------------------------------------------------
__global__ __launch_bounds__(256)
void attn_kernel(const __bf16* __restrict__ Qb,
                 const __bf16* __restrict__ Kb,
                 const __bf16* __restrict__ Vb,
                 const float* __restrict__ offmask,
                 float* __restrict__ attnO,
                 int mode) {
    __shared__ __bf16 sK   [DD][HD];    // K row-major [key][dim]      (16KB)
    __shared__ __bf16 sVrow[DD][HD];    // V rows (TDM landing zone)   (16KB)
    __shared__ __bf16 sVt  [HD][DD];    // V transposed [dim][key]     (16KB)
    __shared__ __bf16 sPs  [8][16][32]; // per-wave P transpose scratch (8KB)

    const int c  = blockIdx.x;   // column index (mode 0) or row index (mode 1)
    const int h  = blockIdx.y;
    const int qh = blockIdx.z;
    const int t    = threadIdx.x;
    const int lane = t & 31;
    const int wave = t >> 5;

    // --- TDM staging: wave 0 issues two 2D tile DMAs (256 rows x 32 bf16). ---
    if (t < 32) {
        const size_t tokK0   = (mode == 0) ? (size_t)c : (size_t)c * DD;
        const unsigned strd  = (mode == 0) ? (unsigned)(DD * COUT) : (unsigned)COUT;
        tdm_load_2d(lds_offset(&sK[0][0]),
                    Kb + tokK0 * COUT + h * HD, DD, HD, strd);
        tdm_load_2d(lds_offset(&sVrow[0][0]),
                    Vb + tokK0 * COUT + h * HD, DD, HD, strd);
        __builtin_amdgcn_s_wait_tensorcnt(0);
    }
    __syncthreads();

    // Transpose V rows -> sVt[dim][key] so PV B-fragments are contiguous.
    {
        V32U vv;
        const uint4* vr = (const uint4*)&sVrow[t][0];
        vv.u4[0] = vr[0]; vv.u4[1] = vr[1]; vv.u4[2] = vr[2]; vv.u4[3] = vr[3];
        #pragma unroll
        for (int d = 0; d < HD; ++d) sVt[d][t] = vv.h[d];
    }
    __syncthreads();

    const int q0 = qh * 128 + wave * 16;
    size_t qtok0; int ldq;
    if (mode == 0) { qtok0 = (size_t)q0 * DD + c; ldq = DD * COUT; }
    else           { qtok0 = (size_t)c  * DD + q0; ldq = COUT;     }

    // Q A-fragment: 16 q-rows x HD(=32) — exactly one K=32 WMMA fragment.
    v16bf aQ = load_frag_a(Qb + qtok0 * COUT + h * HD, ldq, lane);

    // S = Q K^T : 16 N-tiles of 16 keys each.
    v8f acc[16];
    #pragma unroll
    for (int nt = 0; nt < 16; ++nt) acc[nt] = v8f{};
    #pragma unroll
    for (int nt = 0; nt < 16; ++nt) {
        v16bf bK = load_frag_bT(&sK[nt * 16][0], HD, lane);
        acc[nt] = wmma_bf16(aQ, bK, acc[nt]);
    }

    // Scale + off-diagonal mask (faithful: read the mask input).
    const float inv_scale = 0.1767766952966369f;  // 1/sqrt(HD)
    #pragma unroll
    for (int nt = 0; nt < 16; ++nt) {
        const int k = nt * 16 + (lane & 15);
        float mv = (mode == 0) ? offmask[(size_t)k * DD + c]
                               : offmask[(size_t)c * DD + k];
        float neg = (mv == 0.0f) ? -__builtin_inff() : 0.0f;
        #pragma unroll
        for (int g = 0; g < 8; ++g)
            acc[nt][g] = acc[nt][g] * inv_scale + neg;
    }

    // Softmax over k. Row (g, lane-half) data lives across 16 lanes -> width-16
    // xor reductions keep the two halves (m=g and m=g+8) independent.
    float rinv[8];
    #pragma unroll
    for (int g = 0; g < 8; ++g) {
        float m = -__builtin_inff();
        #pragma unroll
        for (int nt = 0; nt < 16; ++nt) m = fmaxf(m, acc[nt][g]);
        #pragma unroll
        for (int s = 1; s < 16; s <<= 1) m = fmaxf(m, __shfl_xor(m, s, 16));
        float sum = 0.0f;
        #pragma unroll
        for (int nt = 0; nt < 16; ++nt) {
            float e = __expf(acc[nt][g] - m);
            acc[nt][g] = e;
            sum += e;
        }
        #pragma unroll
        for (int s = 1; s < 16; s <<= 1) sum += __shfl_xor(sum, s, 16);
        rinv[g] = 1.0f / sum;
    }

    // out = P @ V : K=256 in 8 steps of 32; transpose P through per-wave LDS.
    v8f oacc[2];
    oacc[0] = v8f{}; oacc[1] = v8f{};
    const int halfm = (lane < 16) ? 0 : 8;
    const int nlo   = lane & 15;
    #pragma unroll
    for (int kt = 0; kt < 8; ++kt) {
        #pragma unroll
        for (int j = 0; j < 2; ++j) {
            const int nt = 2 * kt + j;
            #pragma unroll
            for (int g = 0; g < 8; ++g)
                sPs[wave][g + halfm][j * 16 + nlo] = (__bf16)(acc[nt][g] * rinv[g]);
        }
        v16bf aP = load_frag_a(&sPs[wave][0][0], 32, lane);
        #pragma unroll
        for (int v = 0; v < 2; ++v) {
            v16bf bV = load_frag_bT(&sVt[v * 16][kt * 32], DD, lane);
            oacc[v] = wmma_bf16(aP, bV, oacc[v]);
        }
    }

    // Write 0.5*(col) then accumulate 0.5*(row).
    #pragma unroll
    for (int v = 0; v < 2; ++v) {
        const int dim = h * HD + v * 16 + nlo;
        #pragma unroll
        for (int g = 0; g < 8; ++g) {
            const int q = q0 + g + halfm;
            size_t tok = (mode == 0) ? ((size_t)q * DD + c) : ((size_t)c * DD + q);
            float val = 0.5f * oacc[v][g];
            size_t idx = tok * COUT + dim;
            if (mode == 0) attnO[idx] = val;
            else           attnO[idx] = attnO[idx] + val;
        }
    }
}

// ---------------------------------------------------------------------------
// Kernel 3: out = LN( attnO @ Wo + bo + edge_feat ) * gamma + beta, * mask.
// grid = NTOK/128, block = 256. LDS: staging (Wo^T + A) aliased with C tile.
// ---------------------------------------------------------------------------
__global__ __launch_bounds__(256)
void out_proj_kernel(const float* __restrict__ attnO,
                     const float* __restrict__ ef,
                     const float* __restrict__ offmask,
                     const float* __restrict__ Wo,
                     const float* __restrict__ bo,
                     const float* __restrict__ gamma,
                     const float* __restrict__ beta,
                     float* __restrict__ out) {
    __shared__ __align__(16) unsigned char smem[64 * 1024];
    __bf16 (*sWt)[COUT] = (__bf16(*)[COUT])smem;               // 32KB
    __bf16 (*sA)[COUT]  = (__bf16(*)[COUT])(smem + 32768);     // 32KB
    float  (*sC)[COUT]  = (float (*)[COUT])smem;               // 64KB (after sync)

    const int    t    = threadIdx.x;
    const size_t tok0 = (size_t)blockIdx.x * 128;

    {
        const int r  = t >> 1;
        const int c0 = (t & 1) * 64;
        const float* wsrc = Wo + r * COUT + c0;
        #pragma unroll
        for (int j = 0; j < 64; j += 4) {
            float4 f = *(const float4*)(wsrc + j);
            sWt[c0 + j + 0][r] = (__bf16)f.x;
            sWt[c0 + j + 1][r] = (__bf16)f.y;
            sWt[c0 + j + 2][r] = (__bf16)f.z;
            sWt[c0 + j + 3][r] = (__bf16)f.w;
        }
        const float* asrc = attnO + (tok0 + r) * COUT + c0;
        #pragma unroll
        for (int j = 0; j < 64; j += 4) {
            float4 f = *(const float4*)(asrc + j);
            sA[r][c0 + j + 0] = (__bf16)f.x;
            sA[r][c0 + j + 1] = (__bf16)f.y;
            sA[r][c0 + j + 2] = (__bf16)f.z;
            sA[r][c0 + j + 3] = (__bf16)f.w;
        }
    }
    __syncthreads();

    const int lane = t & 31;
    const int wave = t >> 5;
    const int half = (lane < 16) ? 0 : 8;
    const int nlo  = lane & 15;

    v8f acc[8];
    #pragma unroll
    for (int nt = 0; nt < 8; ++nt) acc[nt] = v8f{};
    #pragma unroll
    for (int nt = 0; nt < 8; ++nt) {
        #pragma unroll
        for (int kt = 0; kt < 4; ++kt) {
            v16bf a = load_frag_a (&sA [wave * 16][kt * 32], COUT, lane);
            v16bf b = load_frag_bT(&sWt[nt * 16 ][kt * 32], COUT, lane);
            acc[nt] = wmma_bf16(a, b, acc[nt]);
        }
    }
    __syncthreads();   // everyone done reading sA/sWt; safe to alias as sC

    #pragma unroll
    for (int nt = 0; nt < 8; ++nt) {
        #pragma unroll
        for (int g = 0; g < 8; ++g)
            sC[wave * 16 + g + half][nt * 16 + nlo] = acc[nt][g];
    }
    __syncthreads();

    // Epilogue: one thread per token row (threads 128..255 idle here).
    if (t < 128) {
        const size_t token = tok0 + t;
        const float* efr = ef + token * COUT;
        float sum = 0.0f, sumsq = 0.0f;
        #pragma unroll
        for (int j = 0; j < COUT; j += 4) {
            float4 cvec = *(const float4*)&sC[t][j];
            float4 bvec = *(const float4*)(bo + j);
            float4 evec = *(const float4*)(efr + j);
            float v0 = cvec.x + bvec.x + evec.x;
            float v1 = cvec.y + bvec.y + evec.y;
            float v2 = cvec.z + bvec.z + evec.z;
            float v3 = cvec.w + bvec.w + evec.w;
            sum   += v0 + v1 + v2 + v3;
            sumsq += v0 * v0 + v1 * v1 + v2 * v2 + v3 * v3;
        }
        const float mu  = sum * (1.0f / COUT);
        const float var = sumsq * (1.0f / COUT) - mu * mu;
        const float rsd = rsqrtf(var + 1e-5f);
        const float mv  = offmask[token];
        float* dst = out + token * COUT;
        #pragma unroll
        for (int j = 0; j < COUT; j += 4) {
            float4 cvec = *(const float4*)&sC[t][j];
            float4 bvec = *(const float4*)(bo + j);
            float4 evec = *(const float4*)(efr + j);
            float4 gvec = *(const float4*)(gamma + j);
            float4 zvec = *(const float4*)(beta + j);
            float4 o;
            o.x = (((cvec.x + bvec.x + evec.x) - mu) * rsd * gvec.x + zvec.x) * mv;
            o.y = (((cvec.y + bvec.y + evec.y) - mu) * rsd * gvec.y + zvec.y) * mv;
            o.z = (((cvec.z + bvec.z + evec.z) - mu) * rsd * gvec.z + zvec.z) * mv;
            o.w = (((cvec.w + bvec.w + evec.w) - mu) * rsd * gvec.w + zvec.w) * mv;
            *(float4*)(dst + j) = o;
        }
    }
}

// ---------------------------------------------------------------------------
// Host launcher. Workspace layout (bytes):
//   [0,16M)   Q bf16   [16M,32M) K bf16   [32M,48M) V bf16
//   [48M,80M) attention output fp32 (NTOK x 128)
// ---------------------------------------------------------------------------
extern "C" void kernel_launch(void* const* d_in, const int* in_sizes, int n_in,
                              void* d_out, int out_size, void* d_ws, size_t ws_size,
                              hipStream_t stream) {
    const float* ef      = (const float*)d_in[0];
    const float* offmask = (const float*)d_in[1];
    const float* Wq      = (const float*)d_in[2];
    const float* Wk      = (const float*)d_in[3];
    const float* Wv      = (const float*)d_in[4];
    const float* Wo      = (const float*)d_in[5];
    const float* bo      = (const float*)d_in[6];
    const float* gamma   = (const float*)d_in[7];
    const float* beta    = (const float*)d_in[8];
    float* out = (float*)d_out;

    char* ws = (char*)d_ws;
    const size_t QKV_BYTES = (size_t)NTOK * COUT * sizeof(__bf16);  // 16 MB
    __bf16* Qb    = (__bf16*)(ws);
    __bf16* Kb    = (__bf16*)(ws + QKV_BYTES);
    __bf16* Vb    = (__bf16*)(ws + 2 * QKV_BYTES);
    float*  attnO = (float*) (ws + 3 * QKV_BYTES);                  // 32 MB fp32

    qkv_proj_kernel<<<dim3(NTOK / 128, 3), 256, 0, stream>>>(
        ef, Wq, Wk, Wv, Qb, Kb, Vb);

    attn_kernel<<<dim3(DD, NH, 2), 256, 0, stream>>>(
        Qb, Kb, Vb, offmask, attnO, 0);   // column attention: store 0.5*out
    attn_kernel<<<dim3(DD, NH, 2), 256, 0, stream>>>(
        Qb, Kb, Vb, offmask, attnO, 1);   // row attention: += 0.5*out

    out_proj_kernel<<<NTOK / 128, 256, 0, stream>>>(
        attnO, ef, offmask, Wo, bo, gamma, beta, out);
}
